// MultiLatentAttention_26620207300778
// MI455X (gfx1250) — compile-verified, matrix operand order checked by reading, not run on hardware
//
#include <hip/hip_runtime.h>
#include <hip/hip_bf16.h>
#include <math.h>

// ---------------------------------------------------------------------------
// MLA (Multi-Latent Attention) for MI455X / gfx1250.
// All heavy math on V_WMMA_F32_16X16X32_BF16 (wave32), fp32 accumulate.
// GEMMs: LDS double-buffered, register-prefetched K pipeline.
// Attention: flash-style, bf16 intermediates, async-to-LDS staging.
// ---------------------------------------------------------------------------

typedef __attribute__((ext_vector_type(16))) __bf16 v16bf;
typedef __attribute__((ext_vector_type(8)))  __bf16 v8bf;
typedef __attribute__((ext_vector_type(8)))  float  v8f;
typedef __attribute__((ext_vector_type(4)))  float  f4;
typedef __attribute__((ext_vector_type(4)))  int    v4i;

#define B_   2
#define T_   2048
#define C_   2048
#define NH_  16
#define DH_  128
#define DHR_ 64
#define L_   512

#if defined(__gfx1250__) &&                                                    \
    __has_builtin(__builtin_amdgcn_global_load_async_to_lds_b128) &&           \
    __has_builtin(__builtin_amdgcn_s_wait_asynccnt)
#define USE_ASYNC_LDS 1
#else
#define USE_ASYNC_LDS 0
#endif

static __device__ __forceinline__ __bf16 f2bf(float f) { return (__bf16)f; }

static __device__ __forceinline__ v8f wmma_bf16(v16bf a, v16bf b, v8f c) {
  return __builtin_amdgcn_wmma_f32_16x16x32_bf16(
      /*neg_a=*/false, a, /*neg_b=*/false, b,
      /*c_mod=*/(short)0, c, /*reuse_a=*/false, /*reuse_b=*/false);
}

// 16-byte global->LDS copy; async-to-LDS when the toolchain exposes it.
// Builtin signature (from hipcc diagnostic): param0 = v4i __device__* (AS1
// global source), param1 = v4i AS3 LDS dest, then imm offset, imm cpol.
static __device__ __forceinline__ void cp8bf(__bf16* l, const __bf16* g) {
#if USE_ASYNC_LDS
  typedef __attribute__((address_space(1))) v4i* g4p;
  typedef __attribute__((address_space(3))) v4i* l4p;
  __builtin_amdgcn_global_load_async_to_lds_b128((g4p)g, (l4p)l, 0, 0);
#else
  *(v8bf*)l = *(const v8bf*)g;
#endif
}
static __device__ __forceinline__ void staging_fence() {
#if USE_ASYNC_LDS
  __builtin_amdgcn_s_wait_asynccnt(0);
#endif
  __syncthreads();
}

// 16x32 bf16 A/B fragment from LDS, ISA §7.12.2 layout:
// lane<16 holds K {kbase+0..7, kbase+16..23}; lane>=16 holds {8..15, 24..31}.
static __device__ __forceinline__ v16bf ldfrag(const __bf16* base, int row,
                                               int stride, int kbase) {
  const int lane = threadIdx.x & 31;
  const int koff = kbase + ((lane >> 4) << 3);
  const __bf16* p = base + row * stride + koff;
  v8bf lo = *(const v8bf*)(p);
  v8bf hi = *(const v8bf*)(p + 16);
  v16bf r;
#pragma unroll
  for (int i = 0; i < 8; ++i) { r[i] = lo[i]; r[i + 8] = hi[i]; }
  return r;
}

// ---------------------------------------------------------------------------
// GEMM: C[M,N] = A[M,K(slice)] * W[K,N] (+bias). fp32 in, bf16 WMMA.
// OUTMODE 0: fp32 row-major; 1: bf16 row-major; 2: bf16 transposed (C[n][m]).
// 256 threads / 8 waves; 128x128 tile; K-step 32; LDS double buffer with
// register prefetch of the next K tile (one barrier per K-step).
// ---------------------------------------------------------------------------
template <int OUTMODE>
__global__ __launch_bounds__(256) void gemm_kernel(
    const float* __restrict__ A, int lda, int aoff,
    const float* __restrict__ W, const float* __restrict__ bias,
    void* __restrict__ Cout, int M, int N, int K) {
  __shared__ __attribute__((aligned(16))) __bf16 sA[2][128 * 40];
  __shared__ __attribute__((aligned(16))) __bf16 sB[2][128 * 40];

  const int tid  = threadIdx.x;
  const int lane = tid & 31;
  const int wid  = tid >> 5;
  const int wm   = wid & 3;   // 32 rows per wave
  const int wn   = wid >> 2;  // 64 cols per wave
  const int m0   = blockIdx.y * 128;
  const int n0   = blockIdx.x * 128;

  v8f acc[2][4];
#pragma unroll
  for (int i = 0; i < 2; ++i)
#pragma unroll
    for (int j = 0; j < 4; ++j)
#pragma unroll
      for (int e = 0; e < 8; ++e) acc[i][j][e] = 0.0f;

  f4 ar[4], wr[4];
  auto load_tile = [&](int k0) {
#pragma unroll
    for (int it = 0; it < 4; ++it) {
      int idx = tid + it * 256;
      int row = idx >> 3, c4 = (idx & 7) << 2;
      ar[it] = *(const f4*)(A + (size_t)(m0 + row) * lda + aoff + k0 + c4);
    }
#pragma unroll
    for (int it = 0; it < 4; ++it) {
      int idx = tid + it * 256;
      int kk = idx >> 5, c4 = (idx & 31) << 2;
      int n  = n0 + c4;
      f4 z = {0.0f, 0.0f, 0.0f, 0.0f};
      wr[it] = (n < N) ? *(const f4*)(W + (size_t)(k0 + kk) * N + n) : z;
    }
  };
  auto store_tile = [&](int buf) {
#pragma unroll
    for (int it = 0; it < 4; ++it) {
      int idx = tid + it * 256;
      int row = idx >> 3, c4 = (idx & 7) << 2;
      __bf16* d = &sA[buf][row * 40 + c4];
      d[0] = f2bf(ar[it].x); d[1] = f2bf(ar[it].y);
      d[2] = f2bf(ar[it].z); d[3] = f2bf(ar[it].w);
    }
#pragma unroll
    for (int it = 0; it < 4; ++it) {
      int idx = tid + it * 256;
      int kk = idx >> 5, c4 = (idx & 31) << 2;
      sB[buf][(c4 + 0) * 40 + kk] = f2bf(wr[it].x);
      sB[buf][(c4 + 1) * 40 + kk] = f2bf(wr[it].y);
      sB[buf][(c4 + 2) * 40 + kk] = f2bf(wr[it].z);
      sB[buf][(c4 + 3) * 40 + kk] = f2bf(wr[it].w);
    }
  };

  load_tile(0);
  store_tile(0);
  __syncthreads();

  const int nk = K >> 5;
  for (int ki = 0; ki < nk; ++ki) {
    const int cur = ki & 1;
    if (ki + 1 < nk) load_tile((ki + 1) << 5);  // overlap with WMMAs below

    v16bf af[2], bfr[4];
#pragma unroll
    for (int tm = 0; tm < 2; ++tm)
      af[tm] = ldfrag(&sA[cur][0], wm * 32 + tm * 16 + (lane & 15), 40, 0);
#pragma unroll
    for (int tn = 0; tn < 4; ++tn)
      bfr[tn] = ldfrag(&sB[cur][0], wn * 64 + tn * 16 + (lane & 15), 40, 0);
#pragma unroll
    for (int tm = 0; tm < 2; ++tm)
#pragma unroll
      for (int tn = 0; tn < 4; ++tn)
        acc[tm][tn] = wmma_bf16(af[tm], bfr[tn], acc[tm][tn]);

    if (ki + 1 < nk) {
      store_tile(cur ^ 1);
      __syncthreads();
    }
  }

  // Epilogue: C-layout reg r -> row r (lanes 0-15) / r+8 (lanes 16-31).
#pragma unroll
  for (int tm = 0; tm < 2; ++tm)
#pragma unroll
    for (int tn = 0; tn < 4; ++tn)
#pragma unroll
      for (int r = 0; r < 8; ++r) {
        int row = m0 + wm * 32 + tm * 16 + r + ((lane >> 4) << 3);
        int col = n0 + wn * 64 + tn * 16 + (lane & 15);
        if (col < N) {
          float vv = acc[tm][tn][r];
          if (bias) vv += bias[col];
          if (OUTMODE == 0)
            ((float*)Cout)[(size_t)row * N + col] = vv;
          else if (OUTMODE == 1)
            ((__bf16*)Cout)[(size_t)row * N + col] = f2bf(vv);
          else
            ((__bf16*)Cout)[(size_t)col * M + row] = f2bf(vv);
        }
      }
}

// ---------------------------------------------------------------------------
// RoPE in-place on bf16: theta = BASE^(-(d//2)/D), angle=(t+1)*theta.
// ---------------------------------------------------------------------------
__global__ void rope_bf16_kernel(__bf16* __restrict__ X, int rows, int D,
                                 int Tm) {
  int idx = blockIdx.x * blockDim.x + threadIdx.x;
  int half = D >> 1;
  if (idx >= rows * half) return;
  int row = idx / half;
  int p   = idx - row * half;
  int t   = row % Tm;
  float theta = expf(-(float)p * logf(10000.0f) / (float)D);
  float ang   = (float)(t + 1) * theta;
  float s, c;
  sincosf(ang, &s, &c);
  __bf16* e = X + (size_t)row * D + 2 * p;
  float xe = (float)e[0], xo = (float)e[1];
  e[0] = f2bf(xe * c - xo * s);
  e[1] = f2bf(xo * c + xe * s);
}

// ---------------------------------------------------------------------------
// Flash attention: qf=[q|qRt](192) vs kf=[k|kRt](192), causal, then @V.
// Block = 128 threads (4 waves) = 64 query rows of one (b,h); key tiles of 32.
// All staging is 16-byte copies from bf16 buffers (V pre-transposed in global).
// ---------------------------------------------------------------------------
__global__ __launch_bounds__(128) void mla_attn_kernel(
    const __bf16* __restrict__ qB, const __bf16* __restrict__ qR,
    const __bf16* __restrict__ kB, const __bf16* __restrict__ kR,
    const __bf16* __restrict__ vT,  // [NH*DH][B*T]
    float* __restrict__ out) {
  __shared__ __attribute__((aligned(16))) __bf16 sQ[64 * 192];
  __shared__ __attribute__((aligned(16))) __bf16 sK[32 * 192];
  __shared__ __attribute__((aligned(16))) __bf16 sVt[128 * 32];
  __shared__ __attribute__((aligned(16))) __bf16 sP[64 * 32];

  const int tid   = threadIdx.x;
  const int lane  = tid & 31;
  const int w     = tid >> 5;
  const int qt    = blockIdx.x;
  const int bh    = blockIdx.y;
  const int b     = bh >> 4;
  const int h     = bh & 15;
  const int qBase = qt * 64;
  const int M     = B_ * T_;
  const float scale = rsqrtf((float)(DH_ + DHR_));

  // Stage qf tile (64 x 192): 24 16-byte chunks per row.
  for (int idx = tid; idx < 64 * 24; idx += 128) {
    int row = idx / 24, c = idx - row * 24;
    size_t g = (size_t)(b * T_ + qBase + row);
    const __bf16* src = (c < 16)
        ? qB + g * (NH_ * DH_) + h * DH_ + c * 8
        : qR + g * (NH_ * DHR_) + h * DHR_ + (c - 16) * 8;
    cp8bf(&sQ[row * 192 + c * 8], src);
  }
  staging_fence();

  float m[8], l[8];
  v8f o[8];
#pragma unroll
  for (int r = 0; r < 8; ++r) { m[r] = -1e30f; l[r] = 0.0f; }
#pragma unroll
  for (int tn = 0; tn < 8; ++tn)
#pragma unroll
    for (int e = 0; e < 8; ++e) o[tn][e] = 0.0f;

  const int nkt = 2 * qt + 2;
  for (int j = 0; j < nkt; ++j) {
    __syncthreads();  // previous-iteration readers done before overwrite
    // Stage kf tile (32 x 192) and Vt tile (128 x 32), copy-only.
    for (int idx = tid; idx < 32 * 24; idx += 128) {
      int kk = idx / 24, c = idx - kk * 24;
      size_t g = (size_t)(b * T_ + j * 32 + kk);
      const __bf16* src = (c < 16) ? kB + g * (NH_ * DH_) + h * DH_ + c * 8
                                   : kR + g * DHR_ + (c - 16) * 8;
      cp8bf(&sK[kk * 192 + c * 8], src);
    }
    for (int idx = tid; idx < 128 * 4; idx += 128) {
      int dv = idx >> 2, c = idx & 3;
      cp8bf(&sVt[dv * 32 + c * 8],
            vT + (size_t)(h * DH_ + dv) * M + (size_t)(b * T_ + j * 32 + c * 8));
    }
    // Prefetch next key tile rows (global_prefetch_b8).
    if (j + 1 < nkt) {
      size_t gn = (size_t)(b * T_ + (j + 1) * 32);
      if (tid < 32)
        __builtin_prefetch(kB + (gn + tid) * (NH_ * DH_) + h * DH_, 0, 1);
      else if (tid < 64)
        __builtin_prefetch(kR + (gn + (tid - 32)) * DHR_, 0, 1);
      else
        __builtin_prefetch(vT + (size_t)(h * DH_ + 2 * (tid - 64)) * M + gn, 0, 1);
    }
    staging_fence();

    // S = Qf * Kf^T : 16 rows x 32 key cols per wave.
    v8f c0, c1;
#pragma unroll
    for (int e = 0; e < 8; ++e) { c0[e] = 0.0f; c1[e] = 0.0f; }
#pragma unroll
    for (int ks = 0; ks < 6; ++ks) {
      v16bf a  = ldfrag(sQ, w * 16 + (lane & 15), 192, ks * 32);
      v16bf b0 = ldfrag(sK, (lane & 15), 192, ks * 32);
      v16bf b1 = ldfrag(sK, 16 + (lane & 15), 192, ks * 32);
      c0 = wmma_bf16(a, b0, c0);
      c1 = wmma_bf16(a, b1, c1);
    }

    // Online softmax (in-wave: shfl over the 16-lane halves).
    const int col0  = j * 32 + (lane & 15);
    const int col1  = col0 + 16;
    const int rbase = qBase + w * 16 + ((lane >> 4) << 3);
#pragma unroll
    for (int r = 0; r < 8; ++r) {
      int rowi = rbase + r;
      float s0 = (col0 <= rowi) ? c0[r] * scale : -1e30f;
      float s1 = (col1 <= rowi) ? c1[r] * scale : -1e30f;
      float mx = fmaxf(s0, s1);
#pragma unroll
      for (int msk = 8; msk >= 1; msk >>= 1)
        mx = fmaxf(mx, __shfl_xor(mx, msk, 32));
      float mnew  = fmaxf(m[r], mx);
      float alpha = __expf(m[r] - mnew);
      float p0    = __expf(s0 - mnew);
      float p1    = __expf(s1 - mnew);
      float sum   = p0 + p1;
#pragma unroll
      for (int msk = 8; msk >= 1; msk >>= 1) sum += __shfl_xor(sum, msk, 32);
      l[r] = l[r] * alpha + sum;
      m[r] = mnew;
#pragma unroll
      for (int tn = 0; tn < 8; ++tn) o[tn][r] *= alpha;
      int prow = w * 16 + r + ((lane >> 4) << 3);
      sP[prow * 32 + (lane & 15)]      = f2bf(p0);
      sP[prow * 32 + 16 + (lane & 15)] = f2bf(p1);
    }
    __syncthreads();

    // O += P @ V (K = 32 keys, N = 128 head dims).
    v16bf pa = ldfrag(sP, w * 16 + (lane & 15), 32, 0);
#pragma unroll
    for (int tn = 0; tn < 8; ++tn) {
      v16bf bv = ldfrag(sVt, tn * 16 + (lane & 15), 32, 0);
      o[tn] = wmma_bf16(pa, bv, o[tn]);
    }
  }

  // Normalize, store fp32 in (B,T,NH*DH).
#pragma unroll
  for (int tn = 0; tn < 8; ++tn)
#pragma unroll
    for (int r = 0; r < 8; ++r) {
      int rowi = qBase + w * 16 + r + ((lane >> 4) << 3);
      int col  = h * DH_ + tn * 16 + (lane & 15);
      out[(size_t)(b * T_ + rowi) * (NH_ * DH_) + col] = o[tn][r] / l[r];
    }
}

// ---------------------------------------------------------------------------
template <int MODE>
static inline void launch_gemm(const float* A, int lda, int aoff,
                               const float* W, const float* bias, void* C,
                               int M, int N, int K, hipStream_t s) {
  dim3 g((N + 127) / 128, M / 128);
  gemm_kernel<MODE><<<g, 256, 0, s>>>(A, lda, aoff, W, bias, C, M, N, K);
}

extern "C" void kernel_launch(void* const* d_in, const int* in_sizes, int n_in,
                              void* d_out, int out_size, void* d_ws,
                              size_t ws_size, hipStream_t stream) {
  (void)in_sizes; (void)n_in; (void)out_size; (void)ws_size;
  const float* x   = (const float*)d_in[0];
  const float* WA  = (const float*)d_in[1];
  const float* bA  = (const float*)d_in[2];
  const float* WB  = (const float*)d_in[3];
  const float* bB  = (const float*)d_in[4];
  const float* WC  = (const float*)d_in[5];
  const float* bC  = (const float*)d_in[6];
  const float* WUQ = (const float*)d_in[7];
  const float* WUK = (const float*)d_in[8];
  const float* WUV = (const float*)d_in[9];
  const float* WZ  = (const float*)d_in[10];
  const float* bZ  = (const float*)d_in[11];
  float* outp = (float*)d_out;

  const int M = B_ * T_;  // 4096
  char* ws = (char*)d_ws;
  float*  hbuf = (float*)ws;  ws += (size_t)M * 1024 * 4;  // h, fp32
  __bf16* kRb  = (__bf16*)ws; ws += (size_t)M * 64 * 2;    // kRt, bf16
  __bf16* qRb  = (__bf16*)ws; ws += (size_t)M * 1024 * 2;  // qRt, bf16
  __bf16* qb   = (__bf16*)ws; ws += (size_t)M * 2048 * 2;  // q, bf16
  __bf16* kb   = (__bf16*)ws; ws += (size_t)M * 2048 * 2;  // k, bf16
  __bf16* vtb  = (__bf16*)ws; ws += (size_t)M * 2048 * 2;  // v transposed
  float*  ao   = (float*)ws;  ws += (size_t)M * 2048 * 4;  // attn out, fp32

  // h = x @ WA + bA                     (4096 x 2048 x 1024, fp32 out)
  launch_gemm<0>(x, C_, 0, WA, bA, hbuf, M, 2 * L_, C_, stream);
  // kRt = h @ WB + bB ; qRt = h @ WC + bC  (bf16 out)
  launch_gemm<1>(hbuf, 2 * L_, 0, WB, bB, kRb, M, DHR_, 2 * L_, stream);
  launch_gemm<1>(hbuf, 2 * L_, 0, WC, bC, qRb, M, DHR_ * NH_, 2 * L_, stream);
  // rope (qRt roped over the full 1024 dim, per reference)
  { int tot = M * DHR_ / 2;
    rope_bf16_kernel<<<(tot + 255) / 256, 256, 0, stream>>>(kRb, M, DHR_, T_); }
  { int tot = M * (DHR_ * NH_) / 2;
    rope_bf16_kernel<<<(tot + 255) / 256, 256, 0, stream>>>(qRb, M,
                                                            DHR_ * NH_, T_); }
  // q = q_lat @ WUQ ; k = kv_lat @ WUK (bf16) ; v = kv_lat @ WUV (bf16, T)
  launch_gemm<1>(hbuf, 2 * L_, L_, WUQ, nullptr, qb, M, NH_ * DH_, L_, stream);
  launch_gemm<1>(hbuf, 2 * L_, 0,  WUK, nullptr, kb, M, NH_ * DH_, L_, stream);
  launch_gemm<2>(hbuf, 2 * L_, 0,  WUV, nullptr, vtb, M, NH_ * DH_, L_, stream);
  // causal flash attention
  mla_attn_kernel<<<dim3(T_ / 64, B_ * NH_), 128, 0, stream>>>(qb, qRb, kb,
                                                               kRb, vtb, ao);
  // out = ao @ WZ + bZ                  (4096 x 2048 x 2048, fp32 out)
  launch_gemm<0>(ao, NH_ * DH_, 0, WZ, bZ, outp, M, C_, NH_ * DH_, stream);
}
